// Decoder_16063177687726
// MI455X (gfx1250) — matrix operand order repaired and checked by reading, not compiled, not run on hardware
//
#include <hip/hip_runtime.h>
#include <hip/hip_bf16.h>
#include <math.h>

// Problem constants (from the reference)
#define B_    8
#define T_    16
#define S_    400
#define H_    512
#define C_    512
#define E_    128
#define V_    50000
#define OOV_  50
#define VE_   (V_ + OOV_)

// ---------------------------------------------------------------------------
// CDNA5 WMMA types
// ---------------------------------------------------------------------------
typedef __bf16 bf16x16 __attribute__((ext_vector_type(16)));
typedef __bf16 bf16x8  __attribute__((ext_vector_type(8)));
typedef float  f32x8   __attribute__((ext_vector_type(8)));
typedef float  v8f     __attribute__((ext_vector_type(8)));

__device__ __forceinline__ __bf16 f2bf(float f) {
  // round-to-nearest-even fp32 -> bf16
  union { float f; unsigned u; } a; a.f = f;
  unsigned r = (a.u + 0x7FFFu + ((a.u >> 16) & 1u)) >> 16;
  unsigned short s = (unsigned short)r;
  __bf16 b;
  __builtin_memcpy(&b, &s, 2);
  return b;
}

__device__ __forceinline__ float sigm(float x) { return 1.0f / (1.0f + expf(-x)); }

// ---------------------------------------------------------------------------
// WMMA GEMM, pre-converted bf16 operands (hot path, zero cvt in loop):
//   D[M][N] = A[M][K] (row-major bf16, lda) x B^T (Bm row n = col n of B, ldb)
//            + bias[n]
// One wave per 16x16 output tile; 8 waves / block.
// Register layouts per CDNA5 ISA 7.12.2:
//   A 16x32 bf16 : lane(M=lane&15), chunks K = k0+sel*8..+7 and k0+16+sel*8..+7
//   B 32x16 bf16 : lane(N=lane&15), chunks K = k0+sel*8..+7 and k0+16+sel*8..+7
//   D 16x16 f32  : VGPR r -> M = r + sel*8 ; N = lane&15
// ---------------------------------------------------------------------------
__global__ void __launch_bounds__(256)
wmma_gemm_bf16_kernel(const __bf16* __restrict__ A, int lda,
                      const __bf16* __restrict__ Bm, int ldb,
                      const float* __restrict__ bias,
                      float* __restrict__ D, int ldd,
                      int Mtiles, int Ntiles, int K, int Mvalid)
{
  const int wave = threadIdx.x >> 5;
  const int lane = threadIdx.x & 31;
  const int tile = blockIdx.x * 8 + wave;
  if (tile >= Mtiles * Ntiles) return;          // wave-uniform: EXEC stays all-1s
  const int tm  = tile / Ntiles;
  const int tn  = tile % Ntiles;
  const int l15 = lane & 15;
  const int sel = lane >> 4;

  const __bf16* __restrict__ arow = A  + (size_t)(tm * 16 + l15) * lda;
  const __bf16* __restrict__ brow = Bm + (size_t)(tn * 16 + l15) * ldb;

  v8f acc = {};
#pragma unroll 4
  for (int k0 = 0; k0 < K; k0 += 32) {
    bf16x8 a0 = *(const bf16x8*)(arow + k0 + sel * 8);
    bf16x8 a1 = *(const bf16x8*)(arow + k0 + 16 + sel * 8);
    bf16x16 av = __builtin_shufflevector(a0, a1, 0, 1, 2, 3, 4, 5, 6, 7,
                                         8, 9, 10, 11, 12, 13, 14, 15);
    bf16x8 b0 = *(const bf16x8*)(brow + k0 + sel * 8);
    bf16x8 b1 = *(const bf16x8*)(brow + k0 + 16 + sel * 8);
    bf16x16 bv = __builtin_shufflevector(b0, b1, 0, 1, 2, 3, 4, 5, 6, 7,
                                         8, 9, 10, 11, 12, 13, 14, 15);
    acc = __builtin_amdgcn_wmma_f32_16x16x32_bf16(
        false, av, false, bv, (short)0, acc, false, false);
  }

  const int n = tn * 16 + l15;
  const float bb = bias ? bias[n] : 0.0f;
  float* drow = D + (size_t)(tm * 16 + sel * 8) * ldd + n;
  const int mrem = Mvalid - (tm * 16 + sel * 8);
#pragma unroll
  for (int r = 0; r < 8; ++r)
    if (r < mrem) drow[r * ldd] = acc[r] + bb;
}

// ---------------------------------------------------------------------------
// WMMA GEMM with in-register fp32->bf16 conversion (fallback if ws too small)
// ---------------------------------------------------------------------------
__global__ void __launch_bounds__(256)
wmma_gemm_f32_kernel(const float* __restrict__ A, int lda,
                     const float* __restrict__ Bm, int ldb, int boff,
                     const float* __restrict__ bias,
                     float* __restrict__ D, int ldd,
                     int Mtiles, int Ntiles, int K, int Mvalid)
{
  const int wave = threadIdx.x >> 5;
  const int lane = threadIdx.x & 31;
  const int tile = blockIdx.x * 8 + wave;
  if (tile >= Mtiles * Ntiles) return;
  const int tm  = tile / Ntiles;
  const int tn  = tile % Ntiles;
  const int l15 = lane & 15;
  const int sel = lane >> 4;

  const float* __restrict__ arow = A  + (size_t)(tm * 16 + l15) * lda;
  const float* __restrict__ brow = Bm + (size_t)(tn * 16 + l15) * ldb + boff;

  v8f acc = {};
  for (int k0 = 0; k0 < K; k0 += 32) {
    f32x8 a0 = *(const f32x8*)(arow + k0 + sel * 8);
    f32x8 a1 = *(const f32x8*)(arow + k0 + 16 + sel * 8);
    f32x8 b0 = *(const f32x8*)(brow + k0 + sel * 8);
    f32x8 b1 = *(const f32x8*)(brow + k0 + 16 + sel * 8);
    bf16x16 av, bv;
#pragma unroll
    for (int i = 0; i < 8; ++i) {
      av[i]     = f2bf(a0[i]);
      av[8 + i] = f2bf(a1[i]);
      bv[i]     = f2bf(b0[i]);
      bv[8 + i] = f2bf(b1[i]);
    }
    acc = __builtin_amdgcn_wmma_f32_16x16x32_bf16(
        false, av, false, bv, (short)0, acc, false, false);
  }

  const int n = tn * 16 + l15;
  const float bb = bias ? bias[n] : 0.0f;
  float* drow = D + (size_t)(tm * 16 + sel * 8) * ldd + n;
  const int mrem = Mvalid - (tm * 16 + sel * 8);
#pragma unroll
  for (int r = 0; r < 8; ++r)
    if (r < mrem) drow[r * ldd] = acc[r] + bb;
}

// ---------------------------------------------------------------------------
// fp32 -> bf16 conversion kernels (run once per launch, amortized)
// ---------------------------------------------------------------------------
__global__ void cvt_bf16_kernel(const float* __restrict__ in,
                                __bf16* __restrict__ out, int n)
{
  int g = (blockIdx.x * blockDim.x + threadIdx.x) * 4;
  if (g + 3 < n) {
    bf16x8 v;  // use first 4 lanes
    v[0] = f2bf(in[g]); v[1] = f2bf(in[g + 1]);
    v[2] = f2bf(in[g + 2]); v[3] = f2bf(in[g + 3]);
    __bf16* o = out + g;
    o[0] = v[0]; o[1] = v[1]; o[2] = v[2]; o[3] = v[3];
  } else {
    for (int i = g; i < n; ++i) out[i] = f2bf(in[i]);
  }
}

// out[r][c] = in[r*ldin + coff + c]  for r<rows, c<cols  (strided slice -> bf16)
__global__ void cvt_slice_bf16_kernel(const float* __restrict__ in, int ldin, int coff,
                                      __bf16* __restrict__ out, int rows, int cols)
{
  int g = blockIdx.x * blockDim.x + threadIdx.x;
  if (g >= rows * cols) return;
  int r = g / cols, c = g % cols;
  out[g] = f2bf(in[(size_t)r * ldin + coff + c]);
}

// ---------------------------------------------------------------------------
// Small VALU kernels for the recurrent path
// ---------------------------------------------------------------------------
__global__ void zero_kernel(float* p, int n) {
  int g = blockIdx.x * blockDim.x + threadIdx.x;
  if (g < n) p[g] = 0.0f;
}

__global__ void zero_bf16_kernel(__bf16* p, int n) {
  int g = blockIdx.x * blockDim.x + threadIdx.x;
  if (g < n) p[g] = f2bf(0.0f);
}

// x = concat([emb_t, ctx]) @ xctx_w.T + xctx_b     (B x E)
__global__ void step_x_kernel(const float* __restrict__ emb,
                              const float* __restrict__ ctx_in,
                              const float* __restrict__ xctx_w,
                              const float* __restrict__ xctx_b,
                              float* __restrict__ x, int t)
{
  int g = blockIdx.x * blockDim.x + threadIdx.x;
  if (g >= B_ * E_) return;
  int b = g / E_, e = g % E_;
  const float* wr = xctx_w + (size_t)e * (E_ + C_);
  const float* em = emb + ((size_t)b * T_ + t) * E_;
  const float* cx = ctx_in + (size_t)b * C_;
  float acc = xctx_b[e];
  for (int j = 0; j < E_; ++j) acc += em[j] * wr[j];
  for (int j = 0; j < C_; ++j) acc += cx[j] * wr[E_ + j];
  x[g] = acc;
}

// LSTM cell: gates = x@W_ih.T + b_ih + h@W_hh.T + b_hh ; c,h update
__global__ void lstm_kernel(const float* __restrict__ x,
                            const float* __restrict__ h_in,
                            const float* __restrict__ c_in,
                            const float* __restrict__ W_ih,
                            const float* __restrict__ W_hh,
                            const float* __restrict__ b_ih,
                            const float* __restrict__ b_hh,
                            float* __restrict__ h_out,
                            float* __restrict__ c_out)
{
  int g = blockIdx.x * blockDim.x + threadIdx.x;
  if (g >= B_ * H_) return;
  int b = g / H_, hh = g % H_;
  const float* xb = x + (size_t)b * E_;
  const float* hb = h_in + (size_t)b * H_;
  float acc[4];
#pragma unroll
  for (int q = 0; q < 4; ++q) {
    int row = q * H_ + hh;
    float a = b_ih[row] + b_hh[row];
    const float* wi = W_ih + (size_t)row * E_;
    for (int j = 0; j < E_; ++j) a += xb[j] * wi[j];
    const float* wh = W_hh + (size_t)row * H_;
    for (int j = 0; j < H_; ++j) a += hb[j] * wh[j];
    acc[q] = a;
  }
  float cn = sigm(acc[1]) * c_in[g] + sigm(acc[0]) * tanhf(acc[2]);
  float hn = sigm(acc[3]) * tanhf(cn);
  c_out[g] = cn;
  h_out[g] = hn;
}

// h_term = h_new @ energy_w[:, :H].T    (B x H)
__global__ void hterm_kernel(const float* __restrict__ h,
                             const float* __restrict__ energy_w,
                             float* __restrict__ h_term)
{
  int g = blockIdx.x * blockDim.x + threadIdx.x;
  if (g >= B_ * H_) return;
  int b = g / H_, ho = g % H_;
  const float* wr = energy_w + (size_t)ho * (H_ + C_);
  const float* hb = h + (size_t)b * H_;
  float acc = 0.0f;
  for (int j = 0; j < H_; ++j) acc += hb[j] * wr[j];
  h_term[g] = acc;
}

// Attention: energy/softmax/mask/renorm, context vector, coverage update.
__global__ void __launch_bounds__(256)
attn_kernel(const float* __restrict__ h_term,
            const float* __restrict__ enc_proj,   // B*S x H (includes energy_b)
            const float* __restrict__ cov_in,
            const float* __restrict__ cov_w,      // (H,1) -> flat [h]
            const float* __restrict__ vvec,
            const float* __restrict__ masks,
            const float* __restrict__ enc_out,
            float* __restrict__ probs,
            float* __restrict__ ctx_out,
            float* __restrict__ cov_out,
            float* __restrict__ attn_dist_out,
            float* __restrict__ coverages_out,
            int t)
{
  __shared__ float sh[S_];
  __shared__ float red[256];
  const int b = blockIdx.x, tid = threadIdx.x;
  const float* ht = h_term + (size_t)b * H_;

  for (int s = tid; s < S_; s += 256) {
    float cb = cov_in[b * S_ + s];
    const float* ep = enc_proj + ((size_t)b * S_ + s) * H_;
    float acc = 0.0f;
    for (int h = 0; h < H_; ++h)
      acc += vvec[h] * tanhf(ht[h] + ep[h] + cb * cov_w[h]);
    sh[s] = acc;
  }
  __syncthreads();

  float m = -1e30f;
  for (int s = tid; s < S_; s += 256) m = fmaxf(m, sh[s]);
  red[tid] = m; __syncthreads();
  for (int w = 128; w > 0; w >>= 1) {
    if (tid < w) red[tid] = fmaxf(red[tid], red[tid + w]);
    __syncthreads();
  }
  float smax = red[0]; __syncthreads();

  float psum = 0.0f;
  for (int s = tid; s < S_; s += 256) { float e = expf(sh[s] - smax); sh[s] = e; psum += e; }
  red[tid] = psum; __syncthreads();
  for (int w = 128; w > 0; w >>= 1) {
    if (tid < w) red[tid] += red[tid + w];
    __syncthreads();
  }
  float tot = red[0]; __syncthreads();

  float msum = 0.0f;
  for (int s = tid; s < S_; s += 256) {
    float p = sh[s] / tot * masks[b * S_ + s];
    sh[s] = p; msum += p;
  }
  red[tid] = msum; __syncthreads();
  for (int w = 128; w > 0; w >>= 1) {
    if (tid < w) red[tid] += red[tid + w];
    __syncthreads();
  }
  float den = red[0] + 1e-12f; __syncthreads();

  for (int s = tid; s < S_; s += 256) {
    float p = sh[s] / den;
    sh[s] = p;
    probs[b * S_ + s] = p;
    attn_dist_out[((size_t)b * T_ + t) * S_ + s] = p;
    float cn = cov_in[b * S_ + s] + p;
    cov_out[b * S_ + s] = cn;
    coverages_out[((size_t)b * T_ + t) * S_ + s] = cn;
  }
  __syncthreads();

  for (int cc = tid; cc < C_; cc += 256) {
    float acc = 0.0f;
    const float* eo = enc_out + (size_t)b * S_ * C_ + cc;
    for (int s = 0; s < S_; ++s) acc += sh[s] * eo[(size_t)s * C_];
    ctx_out[b * C_ + cc] = acc;
  }
}

// ah = concat([h_new, ctx_new]) @ attnd_w.T + attnd_b   -> rows 0..7 of 16-row buf
__global__ void attnd_kernel(const float* __restrict__ h,
                             const float* __restrict__ ctx,
                             const float* __restrict__ attnd_w,
                             const float* __restrict__ attnd_b,
                             float* __restrict__ ah)
{
  int g = blockIdx.x * blockDim.x + threadIdx.x;
  if (g >= B_ * H_) return;
  int b = g / H_, hh = g % H_;
  const float* wr = attnd_w + (size_t)hh * (H_ + C_);
  const float* hb = h + (size_t)b * H_;
  const float* cb = ctx + (size_t)b * C_;
  float acc = attnd_b[hh];
  for (int j = 0; j < H_; ++j) acc += hb[j] * wr[j];
  for (int j = 0; j < C_; ++j) acc += cb[j] * wr[H_ + j];
  ah[(size_t)b * H_ + hh] = acc;
}

// p_gen = sigmoid(concat([ctx_new, h_new, x]) @ pgen_w.T + pgen_b)
__global__ void pgen_kernel(const float* __restrict__ ctx,
                            const float* __restrict__ h,
                            const float* __restrict__ x,
                            const float* __restrict__ pgen_w,
                            const float* __restrict__ pgen_b,
                            float* __restrict__ pg)
{
  int b = threadIdx.x;
  if (b >= B_) return;
  float acc = pgen_b[0];
  for (int j = 0; j < C_; ++j) acc += ctx[b * C_ + j] * pgen_w[j];
  for (int j = 0; j < H_; ++j) acc += h[b * H_ + j] * pgen_w[C_ + j];
  for (int j = 0; j < E_; ++j) acc += x[b * E_ + j] * pgen_w[C_ + H_ + j];
  pg[b] = 1.0f / (1.0f + expf(-acc));
}

// Vocab softmax * p_gen, append extra_zeros, scatter-add copy distribution.
__global__ void __launch_bounds__(256)
softmax_final_kernel(const float* __restrict__ logits,
                     const float* __restrict__ pg,
                     const float* __restrict__ probs,
                     const int* __restrict__ ext_idx,
                     const float* __restrict__ extra_zeros,
                     float* __restrict__ out_final, int t)
{
  __shared__ float red[256];
  const int b = blockIdx.x, tid = threadIdx.x;
  const float* lg = logits + (size_t)b * V_;

  float m = -1e30f;
  for (int n = tid; n < V_; n += 256) m = fmaxf(m, lg[n]);
  red[tid] = m; __syncthreads();
  for (int w = 128; w > 0; w >>= 1) {
    if (tid < w) red[tid] = fmaxf(red[tid], red[tid + w]);
    __syncthreads();
  }
  float smax = red[0]; __syncthreads();

  float ps = 0.0f;
  for (int n = tid; n < V_; n += 256) ps += expf(lg[n] - smax);
  red[tid] = ps; __syncthreads();
  for (int w = 128; w > 0; w >>= 1) {
    if (tid < w) red[tid] += red[tid + w];
    __syncthreads();
  }
  float inv = pg[b] / red[0];

  float* row = out_final + ((size_t)b * T_ + t) * VE_;
  for (int n = tid; n < V_; n += 256) row[n] = expf(lg[n] - smax) * inv;
  for (int n = V_ + tid; n < VE_; n += 256) row[n] = extra_zeros[b * OOV_ + (n - V_)];
  __syncthreads();

  float cp = 1.0f - pg[b];
  for (int s = tid; s < S_; s += 256)
    atomicAdd(&row[ext_idx[b * S_ + s]], cp * probs[b * S_ + s]);
}

// ---------------------------------------------------------------------------
// Host orchestration
// ---------------------------------------------------------------------------
extern "C" void kernel_launch(void* const* d_in, const int* in_sizes, int n_in,
                              void* d_out, int out_size, void* d_ws, size_t ws_size,
                              hipStream_t stream)
{
  (void)in_sizes; (void)n_in; (void)out_size;

  const float* emb         = (const float*)d_in[0];
  const float* context     = (const float*)d_in[1];
  const float* h0          = (const float*)d_in[2];
  const float* c0          = (const float*)d_in[3];
  const float* enc_out     = (const float*)d_in[4];
  const float* masks       = (const float*)d_in[5];
  const float* extra_zeros = (const float*)d_in[6];
  const int*   ext_idx     = (const int*)d_in[7];
  const float* coverage    = (const float*)d_in[8];
  const float* W_ih        = (const float*)d_in[9];
  const float* W_hh        = (const float*)d_in[10];
  const float* b_ih        = (const float*)d_in[11];
  const float* b_hh        = (const float*)d_in[12];
  const float* cov_w       = (const float*)d_in[13];
  const float* energy_w    = (const float*)d_in[14];
  const float* energy_b    = (const float*)d_in[15];
  const float* vvec        = (const float*)d_in[16];
  const float* xctx_w      = (const float*)d_in[17];
  const float* xctx_b      = (const float*)d_in[18];
  const float* attnd_w     = (const float*)d_in[19];
  const float* attnd_b     = (const float*)d_in[20];
  const float* vocab_w     = (const float*)d_in[21];
  const float* vocab_b     = (const float*)d_in[22];
  const float* pgen_w      = (const float*)d_in[23];
  const float* pgen_b      = (const float*)d_in[24];

  // Workspace carve-out (256B aligned slices)
  char* ws = (char*)d_ws;
  size_t off = 0;
  auto take = [&](size_t bytes) -> void* {
    char* p = ws + off;
    off = (off + bytes + 255) & ~(size_t)255;
    return (void*)p;
  };
  // Base buffers (~8.4 MB)
  float* enc_proj  = (float*)take((size_t)B_ * S_ * H_ * 4);
  float* logits    = (float*)take((size_t)B_ * V_ * 4);
  float* ah        = (float*)take((size_t)16 * H_ * 4);     // 16 rows (8 zero pad)
  float* xbuf      = (float*)take((size_t)B_ * E_ * 4);
  float* h_term    = (float*)take((size_t)B_ * H_ * 4);
  float* hbuf[2]   = { (float*)take((size_t)B_ * H_ * 4), (float*)take((size_t)B_ * H_ * 4) };
  float* cbuf[2]   = { (float*)take((size_t)B_ * H_ * 4), (float*)take((size_t)B_ * H_ * 4) };
  float* ctxbuf[2] = { (float*)take((size_t)B_ * C_ * 4), (float*)take((size_t)B_ * C_ * 4) };
  float* covbuf[2] = { (float*)take((size_t)B_ * S_ * 4), (float*)take((size_t)B_ * S_ * 4) };
  float* probs     = (float*)take((size_t)B_ * S_ * 4);
  float* pg        = (float*)take(256);
  // bf16 pre-converted operands (~55 MB extra); used only if ws is big enough
  __bf16* vocab_bf  = (__bf16*)take((size_t)V_ * H_ * 2);
  __bf16* encw_bf   = (__bf16*)take((size_t)H_ * C_ * 2);
  __bf16* encout_bf = (__bf16*)take((size_t)B_ * S_ * C_ * 2);
  __bf16* ah_bf     = (__bf16*)take((size_t)16 * H_ * 2);
  const bool use_pre = (ws_size >= off);   // ws_size is fixed per harness -> deterministic

  // Output slices (flat, in reference return order)
  float* out = (float*)d_out;
  const size_t off1 = (size_t)B_ * T_ * VE_;       // ctx
  const size_t off2 = off1 + (size_t)B_ * C_;      // h
  const size_t off3 = off2 + (size_t)B_ * H_;      // c
  const size_t off4 = off3 + (size_t)B_ * H_;      // attn_dists
  const size_t off5 = off4 + (size_t)B_ * T_ * S_; // pgens
  const size_t off6 = off5 + (size_t)B_;           // coverages
  float* out_final = out;
  float* out_attn  = out + off4;
  float* out_cov   = out + off6;

  // Init recurrent state (d2d copies are graph-capture safe)
  hipMemcpyAsync(hbuf[0],   h0,       (size_t)B_ * H_ * 4, hipMemcpyDeviceToDevice, stream);
  hipMemcpyAsync(cbuf[0],   c0,       (size_t)B_ * H_ * 4, hipMemcpyDeviceToDevice, stream);
  hipMemcpyAsync(ctxbuf[0], context,  (size_t)B_ * C_ * 4, hipMemcpyDeviceToDevice, stream);
  hipMemcpyAsync(covbuf[0], coverage, (size_t)B_ * S_ * 4, hipMemcpyDeviceToDevice, stream);
  zero_kernel<<<(8 * H_ + 255) / 256, 256, 0, stream>>>(ah + 8 * H_, 8 * H_);

  if (use_pre) {
    // One-time conversions: static GEMM operands -> bf16 (amortized over 16 steps)
    { int n = V_ * H_;
      cvt_bf16_kernel<<<(n / 4 + 255) / 256, 256, 0, stream>>>(vocab_w, vocab_bf, n); }
    cvt_slice_bf16_kernel<<<((H_ * C_) + 255) / 256, 256, 0, stream>>>(
        energy_w, H_ + C_, H_, encw_bf, H_, C_);
    { int n = B_ * S_ * C_;
      cvt_bf16_kernel<<<(n / 4 + 255) / 256, 256, 0, stream>>>(enc_out, encout_bf, n); }
    zero_bf16_kernel<<<(8 * H_ + 255) / 256, 256, 0, stream>>>(ah_bf + 8 * H_, 8 * H_);
  }

  // Hoisted time-invariant attention term:
  // enc_proj[b*S+s][h] = enc_out[b][s][:] . energy_w[h][H:] + energy_b[h]
  {
    const int Mtiles = (B_ * S_) / 16;   // 200
    const int Ntiles = H_ / 16;          // 32
    const int tiles  = Mtiles * Ntiles;  // 6400
    if (use_pre)
      wmma_gemm_bf16_kernel<<<(tiles + 7) / 8, 256, 0, stream>>>(
          encout_bf, C_, encw_bf, C_, energy_b, enc_proj, H_,
          Mtiles, Ntiles, C_, B_ * S_);
    else
      wmma_gemm_f32_kernel<<<(tiles + 7) / 8, 256, 0, stream>>>(
          enc_out, C_, energy_w, H_ + C_, H_, energy_b, enc_proj, H_,
          Mtiles, Ntiles, C_, B_ * S_);
  }

  int cur = 0;
  for (int t = 0; t < T_; ++t) {
    const int nxt = cur ^ 1;

    step_x_kernel<<<(B_ * E_ + 255) / 256, 256, 0, stream>>>(
        emb, ctxbuf[cur], xctx_w, xctx_b, xbuf, t);

    lstm_kernel<<<(B_ * H_ + 255) / 256, 256, 0, stream>>>(
        xbuf, hbuf[cur], cbuf[cur], W_ih, W_hh, b_ih, b_hh, hbuf[nxt], cbuf[nxt]);

    hterm_kernel<<<(B_ * H_ + 255) / 256, 256, 0, stream>>>(
        hbuf[nxt], energy_w, h_term);

    attn_kernel<<<B_, 256, 0, stream>>>(
        h_term, enc_proj, covbuf[cur], cov_w, vvec, masks, enc_out,
        probs, ctxbuf[nxt], covbuf[nxt], out_attn, out_cov, t);

    attnd_kernel<<<(B_ * H_ + 255) / 256, 256, 0, stream>>>(
        hbuf[nxt], ctxbuf[nxt], attnd_w, attnd_b, ah);

    pgen_kernel<<<1, 32, 0, stream>>>(ctxbuf[nxt], hbuf[nxt], xbuf, pgen_w, pgen_b, pg);

    // Dominant matmul: logits = ah @ vocab_w.T + vocab_b  (WMMA, M padded to 16)
    {
      const int Ntiles = V_ / 16;        // 3125
      if (use_pre) {
        cvt_bf16_kernel<<<(16 * H_ / 4 + 255) / 256, 256, 0, stream>>>(ah, ah_bf, 16 * H_);
        wmma_gemm_bf16_kernel<<<(Ntiles + 7) / 8, 256, 0, stream>>>(
            ah_bf, H_, vocab_bf, H_, vocab_b, logits, V_,
            1, Ntiles, H_, B_);
      } else {
        wmma_gemm_f32_kernel<<<(Ntiles + 7) / 8, 256, 0, stream>>>(
            ah, H_, vocab_w, H_, 0, vocab_b, logits, V_,
            1, Ntiles, H_, B_);
      }
    }

    softmax_final_kernel<<<B_, 256, 0, stream>>>(
        logits, pg, probs, ext_idx, extra_zeros, out_final, t);

    cur = nxt;
  }

  // Remaining outputs: ctx, h, c, last p_gen
  hipMemcpyAsync(out + off1, ctxbuf[cur], (size_t)B_ * C_ * 4, hipMemcpyDeviceToDevice, stream);
  hipMemcpyAsync(out + off2, hbuf[cur],   (size_t)B_ * H_ * 4, hipMemcpyDeviceToDevice, stream);
  hipMemcpyAsync(out + off3, cbuf[cur],   (size_t)B_ * H_ * 4, hipMemcpyDeviceToDevice, stream);
  hipMemcpyAsync(out + off5, pg,          (size_t)B_ * 4,      hipMemcpyDeviceToDevice, stream);
}
